// MultiHeadAttention_25692494365033
// MI455X (gfx1250) — matrix operand order repaired and checked by reading, not compiled, not run on hardware
//
#include <hip/hip_runtime.h>
#include <hip/hip_bf16.h>
#include <math.h>

typedef __attribute__((ext_vector_type(16))) _Float16 v16h;
typedef __attribute__((ext_vector_type(8)))  float    v8f;
typedef _Float16 half_t;

#define NHEADS   16
#define HEADDIM  64
#define HID      1024
#define SEQ      2048
#define BATCH    2
#define MROWS    (BATCH * SEQ)          // 4096

// ---------------------------------------------------------------------------
// Kernel 1: projection GEMM  C = X[4096,1024] @ W[1024,1024]
// Output written as f16 in head-major layout [B, NH, L, HD], scaled by `scale`.
// 8 waves per 256-thread block; each wave owns one 16x16 C tile (WG tile 64x32).
// ---------------------------------------------------------------------------
__global__ __launch_bounds__(256)
void mha_proj_kernel(const float* __restrict__ X, const float* __restrict__ W,
                     half_t* __restrict__ Out, float scale)
{
    const int K = HID, N = HID;
    const int wave = threadIdx.x >> 5;
    const int lane = threadIdx.x & 31;
    const int lr   = lane & 15;
    const int hi   = lane >> 4;          // 0: lanes 0-15, 1: lanes 16-31

    const int row0 = blockIdx.x * 64 + (wave >> 1) * 16;
    const int col0 = blockIdx.y * 32 + (wave & 1) * 16;

    v8f acc = {};
    for (int kk = 0; kk < K; kk += 32) {
        v16h a, b;
        // A (16x32 f16): lane holds row (row0+lr); elem j -> k = kk + 16*hi + j
        const float* arow = X + (size_t)(row0 + lr) * K + kk + hi * 16;
        #pragma unroll
        for (int j = 0; j < 8; ++j) {
            a[j]     = (half_t)arow[j];
            a[j + 8] = (half_t)arow[j + 8];
        }
        // B (32x16 f16): lane holds col (col0+lr); elem j -> k = kk + 16*hi + j
        const float* bcol = W + (size_t)(kk + hi * 16) * N + col0 + lr;
        #pragma unroll
        for (int j = 0; j < 16; ++j)
            b[j] = (half_t)bcol[(size_t)j * N];

        acc = __builtin_amdgcn_wmma_f32_16x16x32_f16(false, a, false, b,
                                                     (short)0, acc, false, false);
    }
    // C/D layout: VGPR r -> row r (lo half) / r+8 (hi half), col = lane&15
    #pragma unroll
    for (int r = 0; r < 8; ++r) {
        const int row = row0 + r + hi * 8;      // = b*SEQ + l
        const int col = col0 + lr;              // = h*HEADDIM + d
        const int bb = row >> 11, l = row & (SEQ - 1);
        const int h  = col >> 6,  d = col & (HEADDIM - 1);
        Out[(((size_t)(bb * NHEADS + h) * SEQ) + l) * HEADDIM + d] =
            (half_t)(acc[r] * scale);
    }
}

// ---------------------------------------------------------------------------
// Kernel 2: fused attention per (b, h, 16-row q tile).
// Phase 1: S = Qs @ K^T (WMMA) into LDS [16 x 2048] f32.
// Phase 2: row softmax in LDS; writes attn (f32) to d_out.
// Phase 3: O = P @ V (WMMA), waves 0-3 each own a 16-col slice of head dim.
// Dynamic LDS: 16*2048 f32 S + 16 max + 16 sum + 16*16 reduce = 132,224 B.
// ---------------------------------------------------------------------------
__global__ __launch_bounds__(256)
void mha_attn_kernel(const half_t* __restrict__ Qh, const half_t* __restrict__ Kh,
                     const half_t* __restrict__ Vh,
                     float* __restrict__ attn_out, float* __restrict__ OH)
{
    extern __shared__ float smem[];
    float* sS   = smem;                    // [16][2048]
    float* sMax = smem + 16 * SEQ;         // [16]
    float* sSum = sMax + 16;               // [16]
    float* sRed = sSum + 16;               // [16][16]

    const int wave = threadIdx.x >> 5;
    const int lane = threadIdx.x & 31;
    const int lr   = lane & 15;
    const int hi   = lane >> 4;

    const int bh = blockIdx.y;             // b*NHEADS + h
    const int q0 = blockIdx.x * 16;

    const half_t* Qb = Qh + (size_t)bh * SEQ * HEADDIM;
    const half_t* Kb = Kh + (size_t)bh * SEQ * HEADDIM;
    const half_t* Vb = Vh + (size_t)bh * SEQ * HEADDIM;

    // ---- Phase 1: scores -------------------------------------------------
    // A tiles (Q rows q0..q0+15, k-dim = head dim 0..63): two 16x32 slabs.
    v16h a0, a1;
    {
        const half_t* arow = Qb + (size_t)(q0 + lr) * HEADDIM + hi * 16;
        #pragma unroll
        for (int j = 0; j < 8; ++j) {
            a0[j]     = arow[j];       a0[j + 8] = arow[j + 8];
            a1[j]     = arow[32 + j];  a1[j + 8] = arow[32 + j + 8];
        }
    }
    for (int kt = wave; kt < SEQ / 16; kt += 8) {
        const int kc = kt * 16;
        v16h b0, b1;
        // B[k][n]: n-th col = key row (kc+lr); contiguous 16 halves per slab
        const half_t* brow = Kb + (size_t)(kc + lr) * HEADDIM + hi * 16;
        #pragma unroll
        for (int j = 0; j < 16; ++j) { b0[j] = brow[j]; b1[j] = brow[32 + j]; }
        v8f s = {};
        s = __builtin_amdgcn_wmma_f32_16x16x32_f16(false, a0, false, b0, (short)0, s, false, false);
        s = __builtin_amdgcn_wmma_f32_16x16x32_f16(false, a1, false, b1, (short)0, s, false, false);
        #pragma unroll
        for (int r = 0; r < 8; ++r)
            sS[(r + hi * 8) * SEQ + kc + lr] = s[r];
    }
    __syncthreads();

    // ---- Phase 2: softmax (16 threads per row, stride-16 columns) --------
    const int row = threadIdx.x >> 4;      // 0..15
    const int sub = threadIdx.x & 15;      // 0..15
    float* Srow = sS + row * SEQ;

    float m = -INFINITY;
    for (int c = sub; c < SEQ; c += 16) m = fmaxf(m, Srow[c]);
    sRed[row * 16 + sub] = m;
    __syncthreads();
    if (sub == 0) {
        float mm = -INFINITY;
        #pragma unroll
        for (int i = 0; i < 16; ++i) mm = fmaxf(mm, sRed[row * 16 + i]);
        sMax[row] = mm;
    }
    __syncthreads();
    const float rm = sMax[row];
    float sum = 0.f;
    for (int c = sub; c < SEQ; c += 16) {
        const float e = __expf(Srow[c] - rm);
        Srow[c] = e;
        sum += e;
    }
    sRed[row * 16 + sub] = sum;
    __syncthreads();
    if (sub == 0) {
        float t = 0.f;
        #pragma unroll
        for (int i = 0; i < 16; ++i) t += sRed[row * 16 + i];
        sSum[row] = t;
    }
    __syncthreads();
    const float inv = 1.0f / sSum[row];
    float* arow_out = attn_out + ((size_t)bh * SEQ + (q0 + row)) * SEQ;
    for (int c = sub; c < SEQ; c += 16) {
        const float p = Srow[c] * inv;
        Srow[c]     = p;
        arow_out[c] = p;
    }
    __syncthreads();

    // ---- Phase 3: O = P @ V (waves 0-3, each a 16-wide head-dim slice) ---
    if (wave < 4) {
        const int n0 = wave * 16;
        v8f o = {};
        for (int kk = 0; kk < SEQ; kk += 32) {
            v16h a, b;
            const float* prow = sS + lr * SEQ + kk + hi * 16;
            #pragma unroll
            for (int j = 0; j < 8; ++j) {
                a[j]     = (half_t)prow[j];
                a[j + 8] = (half_t)prow[j + 8];
            }
            const half_t* bcol = Vb + (size_t)(kk + hi * 16) * HEADDIM + n0 + lr;
            #pragma unroll
            for (int j = 0; j < 16; ++j) b[j] = bcol[(size_t)j * HEADDIM];
            o = __builtin_amdgcn_wmma_f32_16x16x32_f16(false, a, false, b,
                                                       (short)0, o, false, false);
        }
        const int bb = bh >> 4, h = bh & 15;
        #pragma unroll
        for (int r = 0; r < 8; ++r) {
            const int qrow = q0 + r + hi * 8;
            OH[((size_t)(bb * SEQ + qrow)) * HID + h * HEADDIM + n0 + lr] = o[r];
        }
    }
}

// ---------------------------------------------------------------------------
// Kernel 3: output projection + residual.  Xout = resid + OH @ Wfc
// ---------------------------------------------------------------------------
__global__ __launch_bounds__(256)
void mha_outproj_kernel(const float* __restrict__ OH, const float* __restrict__ Wfc,
                        const float* __restrict__ resid, float* __restrict__ Xout)
{
    const int K = HID, N = HID;
    const int wave = threadIdx.x >> 5;
    const int lane = threadIdx.x & 31;
    const int lr   = lane & 15;
    const int hi   = lane >> 4;

    const int row0 = blockIdx.x * 64 + (wave >> 1) * 16;
    const int col0 = blockIdx.y * 32 + (wave & 1) * 16;

    v8f acc = {};
    for (int kk = 0; kk < K; kk += 32) {
        v16h a, b;
        const float* arow = OH + (size_t)(row0 + lr) * K + kk + hi * 16;
        #pragma unroll
        for (int j = 0; j < 8; ++j) {
            a[j]     = (half_t)arow[j];
            a[j + 8] = (half_t)arow[j + 8];
        }
        const float* bcol = Wfc + (size_t)(kk + hi * 16) * N + col0 + lr;
        #pragma unroll
        for (int j = 0; j < 16; ++j)
            b[j] = (half_t)bcol[(size_t)j * N];
        acc = __builtin_amdgcn_wmma_f32_16x16x32_f16(false, a, false, b,
                                                     (short)0, acc, false, false);
    }
    #pragma unroll
    for (int r = 0; r < 8; ++r) {
        const int rowi = row0 + r + hi * 8;
        const int coli = col0 + lr;
        const size_t idx = (size_t)rowi * N + coli;
        Xout[idx] = resid[idx] + acc[r];
    }
}

// ---------------------------------------------------------------------------
// Kernel 4: LayerNorm, one block per row of 1024.
// ---------------------------------------------------------------------------
__global__ __launch_bounds__(256)
void mha_ln_kernel(const float* __restrict__ X, const float* __restrict__ gamma,
                   const float* __restrict__ beta, float* __restrict__ Y)
{
    __shared__ float sbuf[256], sbuf2[256];
    const int row = blockIdx.x;
    const int tid = threadIdx.x;
    const float* xr = X + (size_t)row * HID;

    float v[4], s = 0.f, s2 = 0.f;
    #pragma unroll
    for (int i = 0; i < 4; ++i) {
        v[i] = xr[tid + 256 * i];
        s  += v[i];
        s2 += v[i] * v[i];
    }
    sbuf[tid] = s; sbuf2[tid] = s2;
    __syncthreads();
    for (int off = 128; off > 0; off >>= 1) {
        if (tid < off) { sbuf[tid] += sbuf[tid + off]; sbuf2[tid] += sbuf2[tid + off]; }
        __syncthreads();
    }
    const float mean = sbuf[0] * (1.0f / HID);
    const float var  = sbuf2[0] * (1.0f / HID) - mean * mean;
    const float inv  = rsqrtf(var + 1e-6f);
    #pragma unroll
    for (int i = 0; i < 4; ++i) {
        const int c = tid + 256 * i;
        Y[(size_t)row * HID + c] = (v[i] - mean) * inv * gamma[c] + beta[c];
    }
}

// ---------------------------------------------------------------------------
extern "C" void kernel_launch(void* const* d_in, const int* in_sizes, int n_in,
                              void* d_out, int out_size, void* d_ws, size_t ws_size,
                              hipStream_t stream)
{
    const float* q     = (const float*)d_in[0];
    const float* k     = (const float*)d_in[1];
    const float* v     = (const float*)d_in[2];
    const float* w_qs  = (const float*)d_in[3];
    const float* w_ks  = (const float*)d_in[4];
    const float* w_vs  = (const float*)d_in[5];
    const float* w_fc  = (const float*)d_in[6];
    const float* gamma = (const float*)d_in[7];
    const float* beta  = (const float*)d_in[8];

    float* y    = (float*)d_out;                             // [B, L, HID]
    float* attn = y + (size_t)MROWS * HID;                   // [B, NH, L, L]

    // workspace layout
    char* ws = (char*)d_ws;
    half_t* qh = (half_t*)(ws);                              //  8 MB f16 head-major
    half_t* kh = (half_t*)(ws + ((size_t)8  << 20));         //  8 MB
    half_t* vh = (half_t*)(ws + ((size_t)16 << 20));         //  8 MB
    float*  oh = (float*) (ws + ((size_t)24 << 20));         // 16 MB concat-head out
    float*  x  = (float*) (ws + ((size_t)40 << 20));         // 16 MB pre-LN

    const dim3 blk(256);
    const dim3 gproj(MROWS / 64, HID / 32);                  // 64 x 32 WGs

    mha_proj_kernel<<<gproj, blk, 0, stream>>>(q, w_qs, qh, 0.125f);  // 1/sqrt(64)
    mha_proj_kernel<<<gproj, blk, 0, stream>>>(k, w_ks, kh, 1.0f);
    mha_proj_kernel<<<gproj, blk, 0, stream>>>(v, w_vs, vh, 1.0f);

    const size_t smem = (size_t)(16 * SEQ + 16 + 16 + 16 * 16) * sizeof(float);
    (void)hipFuncSetAttribute((const void*)mha_attn_kernel,
                              hipFuncAttributeMaxDynamicSharedMemorySize, (int)smem);
    mha_attn_kernel<<<dim3(SEQ / 16, BATCH * NHEADS), blk, smem, stream>>>(
        qh, kh, vh, attn, oh);

    mha_outproj_kernel<<<gproj, blk, 0, stream>>>(oh, w_fc, q, x);
    mha_ln_kernel<<<MROWS, blk, 0, stream>>>(x, gamma, beta, y);
}
